// Generator_75814762709323
// MI455X (gfx1250) — matrix operand order repaired and checked by reading, not compiled, not run on hardware
//
#include <hip/hip_runtime.h>
#include <stdint.h>

// ---------------------------------------------------------------------------
// MI455X (gfx1250) persistent bf16-WMMA LSTM generator.
// 32 WGs x 256 threads; wave32; v_wmma_f32_16x16x32_bf16 for all GEMMs.
// All GEMM loads use uniform-base + 32-bit lane offset (saddr addressing)
// to keep VGPR pressure low (no spills, <256 VGPRs).
// ---------------------------------------------------------------------------

typedef __attribute__((ext_vector_type(16))) __bf16 v16bf;
typedef __attribute__((ext_vector_type(8)))  float  v8f;
typedef unsigned short ushort_t;

#define NWG   32
#define NTHR  256
#define BB    128
#define HH    1024
#define LL    256
#define CC    20
#define MM    3

union ABf { uint4 q[2]; v16bf v; };

__device__ __forceinline__ ushort_t f2bf(float f) {
  unsigned u = __builtin_bit_cast(unsigned, f);
  unsigned r = (u + 0x7FFFu + ((u >> 16) & 1u)) >> 16;   // RNE
  return (ushort_t)r;
}
__device__ __forceinline__ float sigmoidf_(float x) {
  return 1.0f / (1.0f + __expf(-x));
}

// Grid barrier: epoch + arrival counter in device memory (agent scope).
__device__ __forceinline__ void gbar(unsigned* ctrl) {
  __threadfence();
  __syncthreads();
  if (threadIdx.x == 0) {
    unsigned e = __hip_atomic_load(ctrl + 1, __ATOMIC_ACQUIRE, __HIP_MEMORY_SCOPE_AGENT);
    unsigned prev = __hip_atomic_fetch_add(ctrl, 1u, __ATOMIC_ACQ_REL, __HIP_MEMORY_SCOPE_AGENT);
    if (prev == NWG - 1) {
      __hip_atomic_store(ctrl, 0u, __ATOMIC_RELAXED, __HIP_MEMORY_SCOPE_AGENT);
      __hip_atomic_fetch_add(ctrl + 1, 1u, __ATOMIC_RELEASE, __HIP_MEMORY_SCOPE_AGENT);
    } else {
      while (__hip_atomic_load(ctrl + 1, __ATOMIC_ACQUIRE, __HIP_MEMORY_SCOPE_AGENT) == e) {
        __builtin_amdgcn_s_sleep(1);
      }
    }
  }
  __syncthreads();
}

// Accumulate this wave's 2 row tiles x (4 gates x 1 col tile) of
// [128 x 4H-slice] += A[128 x K] * W[4H x K]^T.
// A, W: bf16 row-major; uniform base pointers, divergent 32-bit offsets.
__device__ __forceinline__ void gemm_acc(const ushort_t* __restrict__ A, int lda,
                                         const ushort_t* __restrict__ W, int ldw,
                                         int nchunks, int rb, int l15, bool hi,
                                         int wcolbase, v8f acc[4][2])
{
  int aoff0 = (rb + l15) * lda + (hi ? 16 : 0);         // A K {0..7}/{16..23}
  int aoff1 = aoff0 + 16 * lda;
  int woff0 = (0 * HH + wcolbase + l15) * ldw + (hi ? 16 : 0);
  int woff1 = woff0 + HH * ldw;
  int woff2 = woff1 + HH * ldw;
  int woff3 = woff2 + HH * ldw;
  for (int kc = 0; kc < nchunks; ++kc) {
    ABf a0, a1;
    a0.q[0] = *(const uint4*)(A + aoff0);
    a0.q[1] = *(const uint4*)(A + aoff0 + 8);           // K {8..15}/{24..31}
    a1.q[0] = *(const uint4*)(A + aoff1);
    a1.q[1] = *(const uint4*)(A + aoff1 + 8);
    aoff0 += 32; aoff1 += 32;

    ABf b0, b1, b2, b3;
    b0.q[0] = *(const uint4*)(W + woff0);
    b0.q[1] = *(const uint4*)(W + woff0 + 8);
    b1.q[0] = *(const uint4*)(W + woff1);
    b1.q[1] = *(const uint4*)(W + woff1 + 8);
    b2.q[0] = *(const uint4*)(W + woff2);
    b2.q[1] = *(const uint4*)(W + woff2 + 8);
    b3.q[0] = *(const uint4*)(W + woff3);
    b3.q[1] = *(const uint4*)(W + woff3 + 8);
    woff0 += 32; woff1 += 32; woff2 += 32; woff3 += 32;

    acc[0][0] = __builtin_amdgcn_wmma_f32_16x16x32_bf16(false, a0.v, false, b0.v, (short)0, acc[0][0], false, false);
    acc[0][1] = __builtin_amdgcn_wmma_f32_16x16x32_bf16(false, a1.v, false, b0.v, (short)0, acc[0][1], false, false);
    acc[1][0] = __builtin_amdgcn_wmma_f32_16x16x32_bf16(false, a0.v, false, b1.v, (short)0, acc[1][0], false, false);
    acc[1][1] = __builtin_amdgcn_wmma_f32_16x16x32_bf16(false, a1.v, false, b1.v, (short)0, acc[1][1], false, false);
    acc[2][0] = __builtin_amdgcn_wmma_f32_16x16x32_bf16(false, a0.v, false, b2.v, (short)0, acc[2][0], false, false);
    acc[2][1] = __builtin_amdgcn_wmma_f32_16x16x32_bf16(false, a1.v, false, b2.v, (short)0, acc[2][1], false, false);
    acc[3][0] = __builtin_amdgcn_wmma_f32_16x16x32_bf16(false, a0.v, false, b3.v, (short)0, acc[3][0], false, false);
    acc[3][1] = __builtin_amdgcn_wmma_f32_16x16x32_bf16(false, a1.v, false, b3.v, (short)0, acc[3][1], false, false);
  }
}

// ------------------------------ init kernels -------------------------------

__global__ void k_f2bf(const float* __restrict__ src, ushort_t* __restrict__ dst, int n) {
  int i = blockIdx.x * blockDim.x + threadIdx.x;
  if (i < n) dst[i] = f2bf(src[i]);
}

__global__ void k_f2bf_pad(const float* __restrict__ src, ushort_t* __restrict__ dst,
                           int rows, int sc, int dc) {
  int i = blockIdx.x * blockDim.x + threadIdx.x;
  if (i < rows * dc) {
    int r = i / dc, c = i % dc;
    dst[i] = (c < sc) ? f2bf(src[r * sc + c]) : (ushort_t)0;
  }
}

__global__ void k_x0(const float* __restrict__ cond0, const float* __restrict__ noise,
                     ushort_t* __restrict__ x0) {
  int i = blockIdx.x * blockDim.x + threadIdx.x;
  if (i < BB * 64) {
    int b = i >> 6, c = i & 63;
    float v = (c < CC) ? cond0[b * CC + c]
            : (c < CC + 30) ? noise[b * 30 + (c - CC)] : 0.0f;
    x0[i] = f2bf(v);
  }
}

__global__ void k_h0(const float* __restrict__ hinit, ushort_t* __restrict__ h1p,
                     ushort_t* __restrict__ h2p) {
  int i = blockIdx.x * blockDim.x + threadIdx.x;
  if (i < BB * HH) {
    ushort_t v = f2bf(hinit[i]);
    h1p[i] = v; h2p[i] = v;
  }
}

__global__ void k_ctrl(unsigned* ctrl) { ctrl[0] = 0u; ctrl[1] = 0u; }

// ------------------------------ main kernel --------------------------------

__global__ void __launch_bounds__(NTHR, 1)
gen_main(const float* __restrict__ condition, const float* __restrict__ c_init,
         const float* __restrict__ mask,
         const float* __restrict__ bih0, const float* __restrict__ bhh0,
         const float* __restrict__ bih1, const float* __restrict__ bhh1,
         const float* __restrict__ bih2, const float* __restrict__ bhh2,
         const float* __restrict__ Wfc,  const float* __restrict__ bfc,
         const ushort_t* __restrict__ Whh0b, const ushort_t* __restrict__ Wih1b,
         const ushort_t* __restrict__ Whh1b, const ushort_t* __restrict__ Wih2b,
         const ushort_t* __restrict__ Whh2b, const ushort_t* __restrict__ Wih0b,
         const ushort_t* __restrict__ x0pad, ushort_t* __restrict__ xt,
         ushort_t* __restrict__ h1b, ushort_t* __restrict__ h2b,
         float* __restrict__ out_part, unsigned* __restrict__ ctrl,
         float* __restrict__ out)
{
  __shared__ float c1s[BB][32];
  __shared__ float c2s[BB][32];
  __shared__ float hbuf[BB][32];

  const int wg = blockIdx.x;
  const int hbase = wg * 32;
  const int tid = threadIdx.x;
  const int lane = tid & 31;
  const int wid = tid >> 5;
  const int jj = wid & 1;               // col tile within gate (16 cols each)
  const int rb = (wid >> 1) * 32;       // this wave's 32 rows
  const int l15 = lane & 15;
  const bool hi = lane >= 16;
  const int rowOff = hi ? 8 : 0;        // C/D layout: lanes 16-31 hold M = v+8
  const int colLocal = jj * 16 + l15;   // 0..31 within WG slice
  const int hcol = hbase + colLocal;
  const int wcolbase = hbase + jj * 16;

  // persistent cell state in LDS
  for (int i = tid; i < BB * 32; i += NTHR) {
    int r = i >> 5, c = i & 31;
    float cv = c_init[r * HH + hbase + c];
    c1s[r][c] = cv; c2s[r][c] = cv;
  }
  __syncthreads();

  // t-invariant per-lane bias sums (all gates)
  float bs0[4], bs1[4], bs2[4];
#pragma unroll
  for (int g = 0; g < 4; ++g) {
    bs0[g] = bih0[g * HH + hcol] + bhh0[g * HH + hcol];
    bs1[g] = bih1[g * HH + hcol] + bhh1[g * HH + hcol];
    bs2[g] = bih2[g * HH + hcol] + bhh2[g * HH + hcol];
  }
  const float* wfc0 = Wfc + 0 * HH + hbase;
  const float* wfc1 = Wfc + 1 * HH + hbase;
  const float* wfc2 = Wfc + 2 * HH + hbase;

  const int HB = BB * HH;               // elements per h buffer

  for (int t = 0; t < LL; ++t) {
    const int cur = t & 1, prv = cur ^ 1;

    // ---------------- Phase A: cell0 (t==0) / cell1 ----------------
    v8f acc[4][2];
#pragma unroll
    for (int g = 0; g < 4; ++g) {
      acc[g][0] = (v8f){0,0,0,0,0,0,0,0};
      acc[g][1] = (v8f){0,0,0,0,0,0,0,0};
    }
    if (t == 0) {
      gemm_acc(x0pad, 64, Wih0b, 64, 2, rb, l15, hi, wcolbase, acc);
      gemm_acc(h1b + prv * HB, HH, Whh0b, HH, HH / 32, rb, l15, hi, wcolbase, acc);
    } else {
      gemm_acc(xt, 32, Wih1b, 32, 1, rb, l15, hi, wcolbase, acc);
      gemm_acc(h1b + prv * HB, HH, Whh1b, HH, HH / 32, rb, l15, hi, wcolbase, acc);
    }
    {
      const float bA0 = (t == 0) ? bs0[0] : bs1[0];
      const float bA1 = (t == 0) ? bs0[1] : bs1[1];
      const float bA2 = (t == 0) ? bs0[2] : bs1[2];
      const float bA3 = (t == 0) ? bs0[3] : bs1[3];
      const float* maskt = mask + t * (BB * HH);      // uniform base
      ushort_t* h1cur = h1b + cur * HB;
#pragma unroll
      for (int ri = 0; ri < 2; ++ri) {
#pragma unroll
        for (int v = 0; v < 8; ++v) {
          int row = rb + ri * 16 + rowOff + v;
          float gi = acc[0][ri][v] + bA0;
          float gf = acc[1][ri][v] + bA1;
          float gg = acc[2][ri][v] + bA2;
          float go = acc[3][ri][v] + bA3;
          float cn = sigmoidf_(gf) * c1s[row][colLocal] + sigmoidf_(gi) * tanhf(gg);
          c1s[row][colLocal] = cn;
          float h = sigmoidf_(go) * tanhf(cn);
          h *= maskt[row * HH + hcol];                // dropout
          h1cur[row * HH + hcol] = f2bf(h);
        }
      }
    }
    gbar(ctrl);

    // ---------------- Phase B: cell2 + FC partial ----------------
#pragma unroll
    for (int g = 0; g < 4; ++g) {
      acc[g][0] = (v8f){0,0,0,0,0,0,0,0};
      acc[g][1] = (v8f){0,0,0,0,0,0,0,0};
    }
    gemm_acc(h1b + cur * HB, HH, Wih2b, HH, HH / 32, rb, l15, hi, wcolbase, acc);
    gemm_acc(h2b + prv * HB, HH, Whh2b, HH, HH / 32, rb, l15, hi, wcolbase, acc);
    {
      ushort_t* h2cur = h2b + cur * HB;
#pragma unroll
      for (int ri = 0; ri < 2; ++ri) {
#pragma unroll
        for (int v = 0; v < 8; ++v) {
          int row = rb + ri * 16 + rowOff + v;
          float gi = acc[0][ri][v] + bs2[0];
          float gf = acc[1][ri][v] + bs2[1];
          float gg = acc[2][ri][v] + bs2[2];
          float go = acc[3][ri][v] + bs2[3];
          float cn = sigmoidf_(gf) * c2s[row][colLocal] + sigmoidf_(gi) * tanhf(gg);
          c2s[row][colLocal] = cn;
          float h = sigmoidf_(go) * tanhf(cn);
          hbuf[row][colLocal] = h;
          h2cur[row * HH + hcol] = f2bf(h);
        }
      }
    }
    __syncthreads();
    if (tid < BB) {
      const int b = tid;
      float s0 = 0.f, s1 = 0.f, s2 = 0.f;
#pragma unroll
      for (int c = 0; c < 32; ++c) {
        float hv = hbuf[b][c];
        s0 += hv * wfc0[c]; s1 += hv * wfc1[c]; s2 += hv * wfc2[c];
      }
      float* op = out_part + (wg * BB + b) * 4;
      op[0] = s0; op[1] = s1; op[2] = s2;
    }
    gbar(ctrl);

    // ---------------- combine FC, emit out[t], build x_{t+1} ----------------
    if (wg == 0 && tid < BB) {
      const int b = tid;
      float o0 = bfc[0], o1 = bfc[1], o2 = bfc[2];
      for (int w = 0; w < NWG; ++w) {
        const float* op = out_part + (w * BB + b) * 4;
        o0 += op[0]; o1 += op[1]; o2 += op[2];
      }
      float* dst = out + (t * BB + b) * MM;
      dst[0] = o0; dst[1] = o1; dst[2] = o2;
      if (t + 1 < LL) {
        ushort_t* xr = xt + b * 32;
        const float* cr = condition + ((t + 1) * BB + b) * CC;
#pragma unroll
        for (int c = 0; c < CC; ++c) xr[c] = f2bf(cr[c]);
        xr[20] = f2bf(o0); xr[21] = f2bf(o1); xr[22] = f2bf(o2);
#pragma unroll
        for (int c = 23; c < 32; ++c) xr[c] = 0;
      }
    }
    gbar(ctrl);
  }
}

// ------------------------------- host side ---------------------------------

extern "C" void kernel_launch(void* const* d_in, const int* in_sizes, int n_in,
                              void* d_out, int out_size, void* d_ws, size_t ws_size,
                              hipStream_t stream)
{
  const float* condition = (const float*)d_in[0];
  const float* noise     = (const float*)d_in[1];
  const float* h_init    = (const float*)d_in[2];
  const float* c_init    = (const float*)d_in[3];
  const float* mask      = (const float*)d_in[4];
  const float* Wih0 = (const float*)d_in[5];
  const float* Whh0 = (const float*)d_in[6];
  const float* bih0 = (const float*)d_in[7];
  const float* bhh0 = (const float*)d_in[8];
  const float* Wih1 = (const float*)d_in[9];
  const float* Whh1 = (const float*)d_in[10];
  const float* bih1 = (const float*)d_in[11];
  const float* bhh1 = (const float*)d_in[12];
  const float* Wih2 = (const float*)d_in[13];
  const float* Whh2 = (const float*)d_in[14];
  const float* bih2 = (const float*)d_in[15];
  const float* bhh2 = (const float*)d_in[16];
  const float* Wfc  = (const float*)d_in[17];
  const float* bfc  = (const float*)d_in[18];

  char* ws = (char*)d_ws;
  size_t o = 0;
  auto seg = [&](size_t bytes) -> char* {
    char* p = ws + o;
    o += (bytes + 255) & ~(size_t)255;
    return p;
  };
  const size_t WHH = (size_t)4096 * 1024 * 2;
  ushort_t* Whh0b = (ushort_t*)seg(WHH);
  ushort_t* Wih1b = (ushort_t*)seg((size_t)4096 * 32 * 2);
  ushort_t* Whh1b = (ushort_t*)seg(WHH);
  ushort_t* Wih2b = (ushort_t*)seg(WHH);
  ushort_t* Whh2b = (ushort_t*)seg(WHH);
  ushort_t* Wih0b = (ushort_t*)seg((size_t)4096 * 64 * 2);
  ushort_t* x0pad = (ushort_t*)seg((size_t)BB * 64 * 2);
  ushort_t* xt    = (ushort_t*)seg((size_t)BB * 32 * 2);
  ushort_t* h1b   = (ushort_t*)seg((size_t)2 * BB * HH * 2);
  ushort_t* h2b   = (ushort_t*)seg((size_t)2 * BB * HH * 2);
  float*    opart = (float*)seg((size_t)NWG * BB * 4 * 4);
  unsigned* ctrl  = (unsigned*)seg(256);

  const int nW = 4096 * 1024;
  k_f2bf<<<(nW + 255) / 256, 256, 0, stream>>>(Whh0, Whh0b, nW);
  k_f2bf<<<(nW + 255) / 256, 256, 0, stream>>>(Whh1, Whh1b, nW);
  k_f2bf<<<(nW + 255) / 256, 256, 0, stream>>>(Wih2, Wih2b, nW);
  k_f2bf<<<(nW + 255) / 256, 256, 0, stream>>>(Whh2, Whh2b, nW);
  k_f2bf_pad<<<(4096 * 32 + 255) / 256, 256, 0, stream>>>(Wih1, Wih1b, 4096, CC + MM, 32);
  k_f2bf_pad<<<(4096 * 64 + 255) / 256, 256, 0, stream>>>(Wih0, Wih0b, 4096, CC + 30, 64);
  k_x0<<<(BB * 64 + 255) / 256, 256, 0, stream>>>(condition, noise, x0pad);
  // prev buffers (index 1) hold h_init at t=0
  k_h0<<<(BB * HH + 255) / 256, 256, 0, stream>>>(h_init, h1b + (size_t)BB * HH,
                                                  h2b + (size_t)BB * HH);
  k_ctrl<<<1, 1, 0, stream>>>(ctrl);

  gen_main<<<NWG, NTHR, 0, stream>>>(condition, c_init, mask,
                                     bih0, bhh0, bih1, bhh1, bih2, bhh2,
                                     Wfc, bfc,
                                     Whh0b, Wih1b, Whh1b, Wih2b, Whh2b, Wih0b,
                                     x0pad, xt, h1b, h2b, opart, ctrl,
                                     (float*)d_out);
}
